// CorrFast_31842887533354
// MI455X (gfx1250) — matrix operand order repaired and checked by reading, not compiled, not run on hardware
//
#include <hip/hip_runtime.h>

typedef __attribute__((ext_vector_type(16))) _Float16 v16h;
typedef __attribute__((ext_vector_type(8)))  _Float16 v8h;
typedef __attribute__((ext_vector_type(8)))  float    v8f;

#define CB 4
#define CC 96
#define CH 128
#define CW 160
#define PAD 20
#define NOFF 21                 // offsets per dimension (21x21 = 441)
#define OTOT (NOFF * NOFF)
#define HB 8                    // output rows per block
#define RSA 88                  // feat1 LDS row stride (halves): 44 dw, conflict-free
#define RSB 104                 // feat2 LDS row stride (halves): 52 dw, conflict-free
#define F1ROWS 256              // 8 hl * 2 parity * 16 m
#define F2BASE (F1ROWS * RSA)   // 22528 halves (16B aligned)
#define F2ROWS 96               // 2 parity * 48 compacted columns
#define LDSH (F2BASE + F2ROWS * RSB)   // 32512 halves = 65024 B (< 64 KB)

__device__ __forceinline__ v16h cat8(v8h lo, v8h hi) {
    return __builtin_shufflevector(lo, hi, 0,1,2,3,4,5,6,7,8,9,10,11,12,13,14,15);
}

__global__ __launch_bounds__(384)
void corr_wmma_kernel(const float* __restrict__ f1,
                      const float* __restrict__ f2,
                      float* __restrict__ out,
                      float* __restrict__ dump)
{
    __shared__ __align__(16) _Float16 lds[LDSH];
    const int tid  = threadIdx.x;
    const int lane = tid & 31;
    // wave id forced scalar -> role tests become s-branches, address math SALU
    const int wid  = __builtin_amdgcn_readfirstlane(tid >> 5);   // 0..11
    const int w0   = blockIdx.x * 32;
    const int h0   = blockIdx.y * HB;
    const int b    = blockIdx.z;

    // wave roles: (pixel parity) x (delta window) x (h subset)
    const int par  = wid & 1;
    const int del  = ((wid >> 1) % 3) * 16;   // 0, 16, 32
    const int hset = (wid >= 6) ? 1 : 0;
    const int nn   = lane & 15;
    const int hi   = lane >> 4;

    // ---- stage feat1 tile: 8 rows x 32 px x 96 ch, f32->f16, parity-split ----
    for (int it = 0; it < 64; ++it) {
        int flat = tid + 384 * it;            // 0..24575
        int c    = flat >> 8;
        int rem  = flat & 255;
        int hl   = rem >> 5;
        int wl   = rem & 31;                  // coalesced along w
        float v  = f1[((size_t)(b * CC + c) * CH + (h0 + hl)) * CW + (w0 + wl)];
        int row  = (hl * 2 + (wl & 1)) * 16 + (wl >> 1);
        lds[row * RSA + c] = (_Float16)v;
    }

    // ---- feat2 fill invariants: col fixed per thread, channel steps by 4/iter ----
    const int  c0   = tid / 96;               // 0..3
    const int  col  = tid % 96;               // pw = w0 + col
    const int  gw   = w0 + col - PAD;
    const bool gok  = (gw >= 0) && (gw < CW);
    const int  gwc  = gok ? gw : 0;           // clamped -> loads always safe
    const int  foff = c0 * CH * CW + gwc;     // per-lane, invariant
    _Float16* ldsf2 = &lds[F2BASE + ((col & 1) * 48 + (col >> 1)) * RSB + c0];

    // ---- per-lane store offsets (oi/h invariant); invalid -> dump ----
    int  soff[8];
    bool sval[8];
    #pragma unroll
    for (int r = 0; r < 8; ++r) {
        int m = r + hi * 8;
        int v = del + nn - m;
        int w = w0 + 2 * m + par;
        sval[r] = (v >= 0 && v < NOFF);
        soff[r] = w * OTOT + v;
    }
    float* const dl = dump + lane;

    __syncthreads();

    // ---- loop feat2 rows: i = hl + 2*oi, source row r2 = h0 + i - PAD ----
    for (int i0 = 0; i0 < HB + 2 * (NOFF - 1); i0 += 2) {
        #pragma unroll
        for (int ip = 0; ip < 2; ++ip) {      // compile-time hl parity
            const int  i   = i0 + ip;
            const int  r2  = h0 + i - PAD;
            const bool rok = (r2 >= 0) && (r2 < CH);
            const int  r2c = rok ? r2 : 0;
            const float* f2row = f2 + ((size_t)(b * CC) * CH + r2c) * CW; // uniform

            if (col == 0 && rok)              // light next-row prefetch
                __builtin_prefetch(f2row + foff + CW, 0, 1);

            // fill: 24 x (load@imm-offset, select-zero, cvt, ds_store@imm-offset)
            #pragma unroll
            for (int it = 0; it < 24; ++it) {
                float v = f2row[foff + it * (4 * CH * CW)];
                v = (rok && gok) ? v : 0.0f;
                ldsf2[it * 4] = (_Float16)v;
            }
            __syncthreads();

            // B fragments (shared by both h-pairs of this wave)
            v16h bfrag[3];
            {
                const int brow = par * 48 + del + nn;
                #pragma unroll
                for (int kk = 0; kk < 3; ++kk) {
                    const _Float16* pb = &lds[F2BASE + brow * RSB + kk * 32 + hi * 8];
                    bfrag[kk] = cat8(*(const v8h*)pb, *(const v8h*)(pb + 16));
                }
            }

            #pragma unroll
            for (int k2 = 0; k2 < 2; ++k2) {
                const int hl = 4 * k2 + 2 * hset + ip;
                if (hl <= i && i <= hl + 2 * (NOFF - 1)) {   // wave-uniform (scalar)
                    const int oi = (i - hl) >> 1;
                    const int arow = (hl * 2 + par) * 16 + nn;
                    v8f acc = {};
                    #pragma unroll
                    for (int kk = 0; kk < 3; ++kk) {
                        const _Float16* pa = &lds[arow * RSA + kk * 32 + hi * 8];
                        v16h af = cat8(*(const v8h*)pa, *(const v8h*)(pa + 16));
                        acc = __builtin_amdgcn_wmma_f32_16x16x32_f16(
                            false, af, false, bfrag[kk], (short)0, acc, false, false);
                    }
                    float* obase = out + ((size_t)(b * CH + (h0 + hl)) * CW) * OTOT
                                       + oi * NOFF;
                    #pragma unroll
                    for (int r = 0; r < 8; ++r) {
                        float* p = sval[r] ? (obase + soff[r]) : dl;
                        *p = acc[r];
                    }
                }
            }
            __syncthreads();
        }
    }
}

extern "C" void kernel_launch(void* const* d_in, const int* in_sizes, int n_in,
                              void* d_out, int out_size, void* d_ws, size_t ws_size,
                              hipStream_t stream)
{
    (void)in_sizes; (void)n_in; (void)out_size; (void)ws_size;
    const float* feat1 = (const float*)d_in[0];
    const float* feat2 = (const float*)d_in[1];
    float* out  = (float*)d_out;
    float* dump = (float*)d_ws;           // >=128B scratch for dead-lane stores
    dim3 grid(CW / 32, CH / HB, CB);      // 5 x 16 x 4 = 320 workgroups, 12 waves
    corr_wmma_kernel<<<grid, dim3(384, 1, 1), 0, stream>>>(feat1, feat2, out, dump);
}
